// NearestEmbed_19164144075530
// MI455X (gfx1250) — compile-verified, matrix operand order checked by reading, not run on hardware
//
#include <hip/hip_runtime.h>

// ---------------------------------------------------------------------------
// VQ nearest-embedding for MI455X (gfx1250, wave32, WMMA bf16).
// x:[32,64,64,64] f32, emb:[64,1024] f32.
// out = [ result (B*D*H*W f32) | argmin as f32 (B*H*W) ].
// GEMM xf[N,64] @ emb[64,1024] via v_wmma_f32_16x16x32_bf16 with a two-term
// bf16 hi/lo split (3 WMMAs per K-half) for ~fp32 accuracy.
// B tiles double-buffered in LDS; staged with async global->LDS copies
// (ASYNCcnt) when the builtins are available.
// ---------------------------------------------------------------------------

#define D_DIM      64
#define K_CODES    1024
#define HW         4096        // H*W
#define DHW        262144      // D*H*W
#define N_ROWS     131072      // B*H*W
#define RESULT_ELEMS 8388608   // B*D*H*W
#define M_TILE     128         // rows per workgroup (8 waves x 16)
#define NCOLS_MACRO 128        // codebook columns staged in LDS per iter
#define LDS_ROW_DW 36          // padded dword stride per column (32 dw + 4 pad)

#if defined(__has_builtin)
#if __has_builtin(__builtin_amdgcn_global_load_async_to_lds_b128) && \
    __has_builtin(__builtin_amdgcn_s_wait_asynccnt)
#define USE_ASYNC_LDS 1
#endif
#endif
#ifndef USE_ASYNC_LDS
#define USE_ASYNC_LDS 0
#endif

typedef __bf16 bf16_t;
typedef bf16_t v16bf __attribute__((ext_vector_type(16)));
typedef bf16_t v8bf  __attribute__((ext_vector_type(8)));
typedef float  v8f   __attribute__((ext_vector_type(8)));

#if USE_ASYNC_LDS
// Builtin prototype (from clang diagnostics): first arg is
// `int __attribute__((vector_size(16))) __device__ *` (addrspace(1)),
// second is the LDS-side pointer (addrspace(3)), then imm offset + cpol.
typedef int v4i_gcc __attribute__((vector_size(16)));
typedef __attribute__((address_space(1))) v4i_gcc as1_v4i;
typedef __attribute__((address_space(3))) v4i_gcc as3_v4i;
// 16B async DMA: global -> LDS, no VGPR round-trip, tracked by ASYNCcnt.
#define ASYNC_CP16(gp, lp, off)                                               \
  __builtin_amdgcn_global_load_async_to_lds_b128((as1_v4i*)(gp),              \
                                                 (as3_v4i*)(lp), (off), 0)
#endif

#define WMMA_BF16(A, B, C)                                                    \
  __builtin_amdgcn_wmma_f32_16x16x32_bf16(false, (A), false, (B), (short)0,   \
                                          (C), false, false)

// Build one 16-bit A/B fragment (v16bf) for a given lane-half K offset c:
// elements 0..7  = K [c, c+8) ; elements 8..15 = K [c+16, c+24)
__device__ __forceinline__ v16bf load_frag(const bf16_t* p, int c) {
  v8bf q0 = *(const v8bf*)(p + c);
  v8bf q1 = *(const v8bf*)(p + c + 16);
  return __builtin_shufflevector(q0, q1, 0, 1, 2, 3, 4, 5, 6, 7, 8, 9, 10, 11,
                                 12, 13, 14, 15);
}

// ---------------------------------------------------------------------------
// Prep 1: embT hi/lo [1024][64] bf16 (row-major per code) + ||e_k||^2.
// ---------------------------------------------------------------------------
__global__ void prep_emb_kernel(const float* __restrict__ emb,
                                bf16_t* __restrict__ ehi,
                                bf16_t* __restrict__ elo,
                                float* __restrict__ esq) {
  int c = blockIdx.x * blockDim.x + threadIdx.x;
  if (c >= K_CODES) return;
  float s = 0.0f;
#pragma unroll 4
  for (int d = 0; d < D_DIM; ++d) {
    float v = emb[d * K_CODES + c];   // coalesced across threads
    s += v * v;
    bf16_t h = (bf16_t)v;
    ehi[c * D_DIM + d] = h;
    elo[c * D_DIM + d] = (bf16_t)(v - (float)h);
  }
  esq[c] = s;
}

// ---------------------------------------------------------------------------
// Prep 2: transpose x [B,D,H,W] -> xhi/xlo [N][64] bf16 via LDS tiles.
// ---------------------------------------------------------------------------
__global__ void prep_x_kernel(const float* __restrict__ x,
                              bf16_t* __restrict__ xhi,
                              bf16_t* __restrict__ xlo) {
  __shared__ float tile[D_DIM][33];
  const int b  = blockIdx.x >> 7;          // 128 p-blocks per b
  const int p0 = (blockIdx.x & 127) << 5;  // 32 p per block
  const int t  = threadIdx.x;

  const int pl = t & 31;
  const int dg = t >> 5;
  const float* xb = x + (size_t)b * DHW;
#pragma unroll
  for (int dd = 0; dd < 8; ++dd) {
    int d = dg * 8 + dd;
    tile[d][pl] = xb[(size_t)d * HW + p0 + pl];  // 128B coalesced
  }
  __syncthreads();

  const int rowl = t >> 3;
  const int ch   = t & 7;
  const size_t n = (size_t)b * HW + p0 + rowl;
  v8bf hv, lv;
#pragma unroll
  for (int q = 0; q < 8; ++q) {
    float v = tile[ch * 8 + q][rowl];
    bf16_t h = (bf16_t)v;
    hv[q] = h;
    lv[q] = (bf16_t)(v - (float)h);
  }
  *(v8bf*)(xhi + n * D_DIM + ch * 8) = hv;  // 16B coalesced stores
  *(v8bf*)(xlo + n * D_DIM + ch * 8) = lv;
}

// ---------------------------------------------------------------------------
// Main kernel.
// ---------------------------------------------------------------------------
__launch_bounds__(256, 2)
__global__ void vq_main_kernel(const bf16_t* __restrict__ xhi,
                               const bf16_t* __restrict__ xlo,
                               const bf16_t* __restrict__ ehi,
                               const bf16_t* __restrict__ elo,
                               const float* __restrict__ esq,
                               const float* __restrict__ emb,
                               float* __restrict__ out) {
  __shared__ uint32_t s_bh[2][NCOLS_MACRO * LDS_ROW_DW];
  __shared__ uint32_t s_bl[2][NCOLS_MACRO * LDS_ROW_DW];
  __shared__ int s_idx[M_TILE];

  const int tid   = threadIdx.x;
  const int wave  = tid >> 5;
  const int lane  = tid & 31;
  const int lhalf = lane >> 4;
  const int lcol  = lane & 15;
  const int m0    = blockIdx.x * M_TILE;
  const int row   = m0 + wave * 16 + lcol;
  const int c     = lhalf * 8;

  // Stage NCOLS_MACRO columns of embT hi/lo into LDS buffer `buf`.
  auto stage = [&](int nmac, int buf) {
    const int col_l = tid >> 1;  // 0..127
    const int half  = tid & 1;   // 32 bf16 halves each
    const bf16_t* sh = ehi + (size_t)(nmac + col_l) * D_DIM + half * 32;
    const bf16_t* sl = elo + (size_t)(nmac + col_l) * D_DIM + half * 32;
    uint32_t* dh = &s_bh[buf][col_l * LDS_ROW_DW + half * 16];
    uint32_t* dl = &s_bl[buf][col_l * LDS_ROW_DW + half * 16];
#if USE_ASYNC_LDS
    ASYNC_CP16(sh, dh, 0);
    ASYNC_CP16(sh, dh, 16);
    ASYNC_CP16(sh, dh, 32);
    ASYNC_CP16(sh, dh, 48);
    ASYNC_CP16(sl, dl, 0);
    ASYNC_CP16(sl, dl, 16);
    ASYNC_CP16(sl, dl, 32);
    ASYNC_CP16(sl, dl, 48);
#else
    uint4 th[4], tl[4];
#pragma unroll
    for (int q = 0; q < 4; ++q) {
      th[q] = ((const uint4*)sh)[q];
      tl[q] = ((const uint4*)sl)[q];
    }
#pragma unroll
    for (int q = 0; q < 4; ++q) {
      ((uint4*)dh)[q] = th[q];
      ((uint4*)dl)[q] = tl[q];
    }
#endif
  };

  // Resident A fragments (16 rows x 64 K, hi+lo).
  const bf16_t* ah_p = xhi + (size_t)row * D_DIM;
  const bf16_t* al_p = xlo + (size_t)row * D_DIM;
  const v16bf ah0 = load_frag(ah_p, c);
  const v16bf ah1 = load_frag(ah_p, 32 + c);
  const v16bf al0 = load_frag(al_p, c);
  const v16bf al1 = load_frag(al_p, 32 + c);

  float mval[8];
  int   midx[8];
#pragma unroll
  for (int j = 0; j < 8; ++j) { mval[j] = 3.0e38f; midx[j] = 0; }

  stage(0, 0);
#if USE_ASYNC_LDS
  __builtin_amdgcn_s_wait_asynccnt(0);
#endif
  __syncthreads();

  int buf = 0;
  for (int nmac = 0; nmac < K_CODES; nmac += NCOLS_MACRO) {
    // Prefetch next macro-tile into the other buffer while computing.
    if (nmac + NCOLS_MACRO < K_CODES) stage(nmac + NCOLS_MACRO, buf ^ 1);

    for (int nt = 0; nt < NCOLS_MACRO; nt += 16) {
      const bf16_t* bh_p = (const bf16_t*)&s_bh[buf][(nt + lcol) * LDS_ROW_DW];
      const bf16_t* bl_p = (const bf16_t*)&s_bl[buf][(nt + lcol) * LDS_ROW_DW];
      v16bf bh0 = load_frag(bh_p, c);
      v16bf bh1 = load_frag(bh_p, 32 + c);
      v16bf bl0 = load_frag(bl_p, c);
      v16bf bl1 = load_frag(bl_p, 32 + c);

      v8f acc = {};
      acc = WMMA_BF16(ah0, bh0, acc);   // hi*hi   K 0..31
      acc = WMMA_BF16(ah0, bl0, acc);   // hi*lo
      acc = WMMA_BF16(al0, bh0, acc);   // lo*hi
      acc = WMMA_BF16(ah1, bh1, acc);   // hi*hi   K 32..63
      acc = WMMA_BF16(ah1, bl1, acc);   // hi*lo
      acc = WMMA_BF16(al1, bh1, acc);   // lo*hi

      const int colg = nmac + nt + lcol;
      const float es = esq[colg];
#pragma unroll
      for (int j = 0; j < 8; ++j) {
        float s = es - 2.0f * acc[j];   // squared dist minus ||x||^2
        if (s < mval[j]) { mval[j] = s; midx[j] = colg; }
      }
    }

#if USE_ASYNC_LDS
    __builtin_amdgcn_s_wait_asynccnt(0);
#endif
    __syncthreads();
    buf ^= 1;
  }

  // Butterfly min-reduce over the 16 lanes of each half (slot j = row
  // j + 8*lhalf, columns vary across lanes). Tie-break to lowest index.
#pragma unroll
  for (int j = 0; j < 8; ++j) {
    for (int mm = 1; mm < 16; mm <<= 1) {
      float ov = __shfl_xor(mval[j], mm, 32);
      int   oi = __shfl_xor(midx[j], mm, 32);
      if (ov < mval[j] || (ov == mval[j] && oi < midx[j])) {
        mval[j] = ov;
        midx[j] = oi;
      }
    }
  }
  if (lcol < 8) {
    s_idx[wave * 16 + lhalf * 8 + lcol] = midx[lcol];
  }
  __syncthreads();

  // Gather codebook vectors with transposed, coalesced stores.
  for (int i = tid; i < M_TILE * D_DIM; i += 256) {
    const int nl = i & (M_TILE - 1);
    const int d  = i >> 7;
    const int idx = s_idx[nl];
    const float v = emb[d * K_CODES + idx];
    const int n = m0 + nl;
    const int b = n >> 12;
    const int p = n & (HW - 1);
    out[(size_t)b * DHW + (size_t)d * HW + p] = v;
  }
  for (int i = tid; i < M_TILE; i += 256) {
    const int n = m0 + i;
    out[RESULT_ELEMS + n] = (float)s_idx[i];
  }
}

// ---------------------------------------------------------------------------
extern "C" void kernel_launch(void* const* d_in, const int* in_sizes, int n_in,
                              void* d_out, int out_size, void* d_ws,
                              size_t ws_size, hipStream_t stream) {
  (void)in_sizes; (void)n_in; (void)out_size; (void)ws_size;

  const float* x   = (const float*)d_in[0];  // [32,64,64,64]
  const float* emb = (const float*)d_in[1];  // [64,1024]
  float* out = (float*)d_out;

  char* ws = (char*)d_ws;
  bf16_t* xhi = (bf16_t*)(ws);                           // 16 MiB
  bf16_t* xlo = (bf16_t*)(ws + 16777216);                // 16 MiB
  bf16_t* ehi = (bf16_t*)(ws + 33554432);                // 128 KiB
  bf16_t* elo = (bf16_t*)(ws + 33554432 + 131072);       // 128 KiB
  float*  esq = (float*)(ws + 33554432 + 262144);        // 4 KiB

  prep_emb_kernel<<<4, 256, 0, stream>>>(emb, ehi, elo, esq);
  prep_x_kernel<<<4096, 256, 0, stream>>>(x, xhi, xlo);
  vq_main_kernel<<<N_ROWS / M_TILE, 256, 0, stream>>>(xhi, xlo, ehi, elo, esq,
                                                      emb, out);
}
